// AttentionLayer_10634339025618
// MI455X (gfx1250) — matrix operand order repaired and checked by reading, not compiled, not run on hardware
//
#include <hip/hip_runtime.h>
#include <hip/hip_bf16.h>
#include <hip/hip_fp16.h>

// ---------------------------------------------------------------------------
// AttentionLayer for MI455X (gfx1250): f16 WMMA everywhere, f32 accumulate.
// Pipeline: cvt(x) + transpose-cvt(weights) -> QKV GEMM (fused head split +
// present write + V transpose) -> flash attention (online softmax) -> proj GEMM
// GEMM mainloop: double-buffered LDS, async global->LDS (ASYNCcnt) when the
// toolchain exposes the gfx1250 builtins, else sync b128 staging + prefetch.
// ---------------------------------------------------------------------------

typedef __attribute__((ext_vector_type(16))) _Float16 v16h;
typedef __attribute__((ext_vector_type(8)))  _Float16 v8h;
typedef __attribute__((ext_vector_type(8)))  float    v8f;

#define BB 2
#define SS 2048
#define DD 1024
#define HH 16
#define HD 64
#define MM (BB * SS)          // 4096 rows
#define A_ELEMS ((size_t)MM * DD)
#define KSTEP 64
#define NSTAGE (DD / KSTEP)   // 16

// ---- gfx1250 async global->LDS support (guarded; falls back to sync) ------
#if defined(__has_builtin)
# if __has_builtin(__builtin_amdgcn_global_load_async_to_lds_b128)
#  define USE_ASYNC_LDS 1
# endif
#endif
#ifndef USE_ASYNC_LDS
# define USE_ASYNC_LDS 0
#endif

// Exact builtin operand type per hipcc diagnostic:
//   "__attribute__((__vector_size__(4 * sizeof(int)))) int __device__ *"
typedef int b128i __attribute__((__vector_size__(4 * sizeof(int))));
typedef __attribute__((address_space(1))) b128i gbl_b128;   // __device__ AS
typedef __attribute__((address_space(3))) b128i lds_b128;   // LDS AS

static __device__ inline void async_copy16(const _Float16* g, _Float16* l) {
#if USE_ASYNC_LDS
  // 16B per lane, global -> LDS, tracked by ASYNCcnt.
  __builtin_amdgcn_global_load_async_to_lds_b128(
      (gbl_b128*)(uintptr_t)g,
      (lds_b128*)(unsigned)(uintptr_t)l,   // low 32 bits of generic = LDS off
      0, 0);
#else
  (void)g; (void)l;
#endif
}

static __device__ inline void wait_async0() {
#if defined(__has_builtin) && __has_builtin(__builtin_amdgcn_s_wait_asynccnt)
  __builtin_amdgcn_s_wait_asynccnt(0);
#else
  asm volatile("s_wait_asynccnt 0" ::: "memory");
#endif
}

static __device__ inline v8f wmma_f16(v16h a, v16h b, v8f c) {
  // D = A(16x32,f16) x B(32x16,f16) + C(16x16,f32)
  return __builtin_amdgcn_wmma_f32_16x16x32_f16(false, a, false, b,
                                                (short)0, c, false, false);
}

// A fragment (16x32 f16, MxK). Per ISA 7.12.2: lane<16 row M=lane holds
// K=0..7 (vgpr0-3) and K=16..23 (vgpr4-7); lane>=16 holds K=8..15 / 24..31.
static __device__ inline v16h load_a_frag(const _Float16* base, int ldk) {
  const int lane = threadIdx.x & 31;
  const int row  = lane & 15;
  const int grp  = lane >> 4;
  v8h lo = *(const v8h*)(base + row * ldk + grp * 8);
  v8h hi = *(const v8h*)(base + row * ldk + 16 + grp * 8);
  v16h out;
#pragma unroll
  for (int i = 0; i < 8; ++i) { out[i] = lo[i]; out[i + 8] = hi[i]; }
  return out;
}

// B fragment loaded from a B^T (N x K) row-major buffer. Lanes 0-15 hold
// K=0..7 / 16..23 for column N=lane; lanes 16-31 hold K=8..15 / 24..31.
static __device__ inline v16h load_bt_frag(const _Float16* base, int ldk) {
  const int lane = threadIdx.x & 31;
  const int row  = lane & 15;
  const int grp  = lane >> 4;
  v8h lo = *(const v8h*)(base + row * ldk + grp * 8);
  v8h hi = *(const v8h*)(base + row * ldk + 16 + grp * 8);
  v16h out;
#pragma unroll
  for (int i = 0; i < 8; ++i) { out[i] = lo[i]; out[i + 8] = hi[i]; }
  return out;
}

static __device__ inline v8f zero8() {
  v8f z;
#pragma unroll
  for (int i = 0; i < 8; ++i) z[i] = 0.0f;
  return z;
}

// ---------------------------------------------------------------------------
// Conversion kernels
// ---------------------------------------------------------------------------
__global__ void k_cvt_f16(const float* __restrict__ in,
                          _Float16* __restrict__ out, int n) {
  int i = blockIdx.x * blockDim.x + threadIdx.x;
  int stride = gridDim.x * blockDim.x;
  for (; i < n; i += stride) out[i] = (_Float16)in[i];
}

// out[n*K + k] = (f16) in[k*N + n]   (produce B^T, N-major)
__global__ void k_tcvt_f16(const float* __restrict__ in,
                           _Float16* __restrict__ out, int K, int N) {
  int idx = blockIdx.x * blockDim.x + threadIdx.x;
  if (idx >= K * N) return;
  int n = idx / K, k = idx % K;
  out[idx] = (_Float16)in[(size_t)k * N + n];
}

// ---------------------------------------------------------------------------
// 64x64 GEMM mainloop: C(64x64) += A(64xK) * Bt(64xK)^T, K = 1024.
// 128 threads = 4 waves in 2x2; each wave owns a 32x32 tile (2x2 WMMA frags).
// Double-buffered LDS stages of K=64; async DMA overlap when available.
// ---------------------------------------------------------------------------
static __device__ void gemm64x64(const _Float16* __restrict__ A,
                                 const _Float16* __restrict__ Bt,
                                 int m0, int n0,
                                 _Float16* ldsA, _Float16* ldsB, // each [2][64*64]
                                 v8f acc[2][2]) {
  const int tid  = threadIdx.x;
  const int wave = tid >> 5;
  const int wm   = wave >> 1;
  const int wn   = wave & 1;

  // Per-thread copy coords: 4 chunks of 8 halves per matrix per stage
  // (64 rows x 64 halves = 512 x 16B chunks / 128 threads).
  int rr[4], co[4];
#pragma unroll
  for (int it = 0; it < 4; ++it) {
    int ch = it * 128 + tid;
    rr[it] = ch >> 3;
    co[it] = (ch & 7) * 8;
  }

  auto compute = [&](int buf) {
    const _Float16* lA = ldsA + buf * 4096;
    const _Float16* lB = ldsB + buf * 4096;
#pragma unroll
    for (int kc = 0; kc < 2; ++kc) {
      v16h a0 = load_a_frag (lA + (wm * 32)      * 64 + kc * 32, 64);
      v16h a1 = load_a_frag (lA + (wm * 32 + 16) * 64 + kc * 32, 64);
      v16h b0 = load_bt_frag(lB + (wn * 32)      * 64 + kc * 32, 64);
      v16h b1 = load_bt_frag(lB + (wn * 32 + 16) * 64 + kc * 32, 64);
      acc[0][0] = wmma_f16(a0, b0, acc[0][0]);
      acc[0][1] = wmma_f16(a0, b1, acc[0][1]);
      acc[1][0] = wmma_f16(a1, b0, acc[1][0]);
      acc[1][1] = wmma_f16(a1, b1, acc[1][1]);
    }
  };

#if USE_ASYNC_LDS
  auto stage_async = [&](int kt, int buf) {
#pragma unroll
    for (int it = 0; it < 4; ++it) {
      async_copy16(A  + (size_t)(m0 + rr[it]) * DD + kt + co[it],
                   ldsA + buf * 4096 + rr[it] * 64 + co[it]);
      async_copy16(Bt + (size_t)(n0 + rr[it]) * DD + kt + co[it],
                   ldsB + buf * 4096 + rr[it] * 64 + co[it]);
    }
  };
  stage_async(0, 0);
  wait_async0();
  __syncthreads();
  for (int kt = 0; kt < NSTAGE; ++kt) {
    if (kt + 1 < NSTAGE) stage_async((kt + 1) * KSTEP, (kt + 1) & 1);
    compute(kt & 1);                 // overlaps with in-flight async DMA
    if (kt + 1 < NSTAGE) {
      wait_async0();                 // next buffer resident in LDS
      __syncthreads();
    }
  }
#else
  v8h ra[4], rb[4];
  auto stage_load = [&](int kt) {    // global -> regs (+ prefetch next slab)
#pragma unroll
    for (int it = 0; it < 4; ++it) {
      const _Float16* pa = A  + (size_t)(m0 + rr[it]) * DD + kt + co[it];
      const _Float16* pb = Bt + (size_t)(n0 + rr[it]) * DD + kt + co[it];
      ra[it] = *(const v8h*)pa;
      rb[it] = *(const v8h*)pb;
      if (kt + KSTEP < DD) {
        __builtin_prefetch(pa + KSTEP, 0, 1);   // global_prefetch_b8
        __builtin_prefetch(pb + KSTEP, 0, 1);
      }
    }
  };
  auto stage_store = [&](int buf) {  // regs -> lds
#pragma unroll
    for (int it = 0; it < 4; ++it) {
      *(v8h*)(ldsA + buf * 4096 + rr[it] * 64 + co[it]) = ra[it];
      *(v8h*)(ldsB + buf * 4096 + rr[it] * 64 + co[it]) = rb[it];
    }
  };
  stage_load(0);
  stage_store(0);
  __syncthreads();
  for (int kt = 0; kt < NSTAGE; ++kt) {
    if (kt + 1 < NSTAGE) stage_load((kt + 1) * KSTEP);
    compute(kt & 1);
    if (kt + 1 < NSTAGE) {
      stage_store((kt + 1) & 1);     // other buffer: safe vs. current readers
      __syncthreads();
    }
  }
#endif
}

// ---------------------------------------------------------------------------
// QKV GEMM: [4096 x 3072] = xh @ w_attn + b_attn, fused split/transpose
// epilogue: q -> qh[B,H,S,hd] f16 ; k -> kh f16 + present f32 ;
//           v -> vt[B,H,hd,S] f16 (transposed) + present f32
// ---------------------------------------------------------------------------
__global__ __launch_bounds__(128)
void k_gemm_qkv(const _Float16* __restrict__ xh,
                const _Float16* __restrict__ wat,   // [3D][D] = W^T
                const float* __restrict__ b_attn,
                _Float16* __restrict__ qh, _Float16* __restrict__ kh,
                _Float16* __restrict__ vt, float* __restrict__ present) {
  __shared__ __align__(16) _Float16 ldsA[2][64 * 64];
  __shared__ __align__(16) _Float16 ldsB[2][64 * 64];
  const int m0 = blockIdx.y * 64;
  const int n0 = blockIdx.x * 64;
  v8f acc[2][2];
#pragma unroll
  for (int i = 0; i < 2; ++i)
#pragma unroll
    for (int j = 0; j < 2; ++j) acc[i][j] = zero8();

  gemm64x64(xh, wat, m0, n0, &ldsA[0][0], &ldsB[0][0], acc);

  const int wave = threadIdx.x >> 5;
  const int wm = wave >> 1, wn = wave & 1;
  const int lane = threadIdx.x & 31;
  const int grp = lane >> 4, col = lane & 15;
#pragma unroll
  for (int i = 0; i < 2; ++i)
#pragma unroll
    for (int j = 0; j < 2; ++j)
#pragma unroll
      for (int r = 0; r < 8; ++r) {
        int gm = m0 + wm * 32 + i * 16 + r + grp * 8;   // row in B*S
        int gn = n0 + wn * 32 + j * 16 + col;           // col in 3D
        float val = acc[i][j][r] + b_attn[gn];
        int b = gm >> 11, s = gm & 2047;
        if (gn < DD) {
          int h = gn >> 6, d = gn & 63;
          qh[(((size_t)(b * HH + h)) * SS + s) * HD + d] = (_Float16)val;
        } else if (gn < 2 * DD) {
          int f = gn - DD, h = f >> 6, d = f & 63;
          kh[(((size_t)(b * HH + h)) * SS + s) * HD + d] = (_Float16)val;
          present[(((size_t)((b * 2 + 0) * HH + h)) * SS + s) * HD + d] = val;
        } else {
          int f = gn - 2 * DD, h = f >> 6, d = f & 63;
          vt[(((size_t)(b * HH + h)) * HD + d) * SS + s] = (_Float16)val;
          present[(((size_t)((b * 2 + 1) * HH + h)) * SS + s) * HD + d] = val;
        }
      }
}

// ---------------------------------------------------------------------------
// Flash attention: 1 wave per 16-row query tile; online softmax;
// P re-shaped C-frag -> A-frag through per-wave LDS.
// ---------------------------------------------------------------------------
__global__ __launch_bounds__(128)
void k_attn(const _Float16* __restrict__ qh, const _Float16* __restrict__ kh,
            const _Float16* __restrict__ vt, _Float16* __restrict__ ah) {
  __shared__ __align__(16) _Float16 ldsP[4][16 * 32];
  const int wave = threadIdx.x >> 5;
  const int tile = blockIdx.x * 4 + wave;   // 4096 tiles total
  const int qt = tile & 127;                // S/16
  const int bh = tile >> 7;                 // b*H + h
  const int q0 = qt * 16;
  const int lane = threadIdx.x & 31;
  const int grp = lane >> 4, col = lane & 15;

  const _Float16* qbase = qh + ((size_t)bh * SS + q0) * HD;
  v16h qa0 = load_a_frag(qbase, HD);        // d = 0..31
  v16h qa1 = load_a_frag(qbase + 32, HD);   // d = 32..63

  v8f o[4];
  float rm[8], rs[8];
#pragma unroll
  for (int dt = 0; dt < 4; ++dt) o[dt] = zero8();
#pragma unroll
  for (int r = 0; r < 8; ++r) { rm[r] = -3.0e38f; rs[r] = 0.0f; }

  _Float16* pw = &ldsP[wave][0];

  for (int kb = 0; kb < q0 + 16; kb += 32) {    // causal: keys <= q0+15
    v8f st[2];
#pragma unroll
    for (int t = 0; t < 2; ++t) {
      const _Float16* kbase = kh + ((size_t)bh * SS + kb + t * 16) * HD;
      v16h k0 = load_bt_frag(kbase, HD);
      v16h k1 = load_bt_frag(kbase + 32, HD);
      st[t] = wmma_f16(qa0, k0, zero8());
      st[t] = wmma_f16(qa1, k1, st[t]);
    }
#pragma unroll
    for (int r = 0; r < 8; ++r) {
      const int q = q0 + r + grp * 8;
      float s0 = st[0][r] * 0.125f;             // 1/sqrt(64)
      float s1 = st[1][r] * 0.125f;
      if (kb + col > q)      s0 = -3.0e30f;     // causal mask
      if (kb + 16 + col > q) s1 = -3.0e30f;
      float v = fmaxf(s0, s1);
      v = fmaxf(v, __shfl_xor(v, 1));
      v = fmaxf(v, __shfl_xor(v, 2));
      v = fmaxf(v, __shfl_xor(v, 4));
      v = fmaxf(v, __shfl_xor(v, 8));           // row max within 16-lane group
      float mnew = fmaxf(rm[r], v);
      float corr = __expf(rm[r] - mnew);
      rm[r] = mnew;
      rs[r] *= corr;
#pragma unroll
      for (int dt = 0; dt < 4; ++dt) o[dt][r] *= corr;
      float p0 = __expf(s0 - mnew);
      float p1 = __expf(s1 - mnew);
      rs[r] += p0 + p1;                         // per-lane partial row sum
      int m = r + grp * 8;
      pw[m * 32 + col]      = (_Float16)p0;
      pw[m * 32 + 16 + col] = (_Float16)p1;
    }
    v16h pa = load_a_frag(pw, 32);              // P as 16x32 A fragment
#pragma unroll
    for (int dt = 0; dt < 4; ++dt) {
      v16h vf = load_bt_frag(vt + ((size_t)bh * HD + dt * 16) * SS + kb, SS);
      o[dt] = wmma_f16(pa, vf, o[dt]);
    }
  }

  float inv[8];
#pragma unroll
  for (int r = 0; r < 8; ++r) {
    float s = rs[r];
    s += __shfl_xor(s, 1);
    s += __shfl_xor(s, 2);
    s += __shfl_xor(s, 4);
    s += __shfl_xor(s, 8);
    inv[r] = 1.0f / s;
  }
  const int b = bh >> 4, h = bh & 15;
#pragma unroll
  for (int dt = 0; dt < 4; ++dt)
#pragma unroll
    for (int r = 0; r < 8; ++r) {
      int sidx = q0 + r + grp * 8;
      int d = dt * 16 + col;
      ah[((size_t)b * SS + sidx) * DD + h * HD + d] =
          (_Float16)(o[dt][r] * inv[r]);
    }
}

// ---------------------------------------------------------------------------
// Output projection GEMM: d_out[0..B*S*D) = ah @ w_proj + b_proj  (f32 out)
// ---------------------------------------------------------------------------
__global__ __launch_bounds__(128)
void k_gemm_proj(const _Float16* __restrict__ ah,
                 const _Float16* __restrict__ wpt,  // [D][D] = W^T
                 const float* __restrict__ b_proj,
                 float* __restrict__ out) {
  __shared__ __align__(16) _Float16 ldsA[2][64 * 64];
  __shared__ __align__(16) _Float16 ldsB[2][64 * 64];
  const int m0 = blockIdx.y * 64;
  const int n0 = blockIdx.x * 64;
  v8f acc[2][2];
#pragma unroll
  for (int i = 0; i < 2; ++i)
#pragma unroll
    for (int j = 0; j < 2; ++j) acc[i][j] = zero8();

  gemm64x64(ah, wpt, m0, n0, &ldsA[0][0], &ldsB[0][0], acc);

  const int wave = threadIdx.x >> 5;
  const int wm = wave >> 1, wn = wave & 1;
  const int lane = threadIdx.x & 31;
  const int grp = lane >> 4, col = lane & 15;
#pragma unroll
  for (int i = 0; i < 2; ++i)
#pragma unroll
    for (int j = 0; j < 2; ++j)
#pragma unroll
      for (int r = 0; r < 8; ++r) {
        int gm = m0 + wm * 32 + i * 16 + r + grp * 8;
        int gn = n0 + wn * 32 + j * 16 + col;
        out[(size_t)gm * DD + gn] = acc[i][j][r] + b_proj[gn];
      }
}

// ---------------------------------------------------------------------------
// Launch
// ---------------------------------------------------------------------------
extern "C" void kernel_launch(void* const* d_in, const int* in_sizes, int n_in,
                              void* d_out, int out_size, void* d_ws,
                              size_t ws_size, hipStream_t stream) {
  const float* x      = (const float*)d_in[0];
  const float* w_attn = (const float*)d_in[1];
  const float* b_attn = (const float*)d_in[2];
  const float* w_proj = (const float*)d_in[3];
  const float* b_proj = (const float*)d_in[4];
  float* out = (float*)d_out;

  char* ws = (char*)d_ws;
  _Float16* xh  = (_Float16*)(ws);              //  8 MB  x in f16
  _Float16* wat = (_Float16*)(ws + 8388608);    //  6 MB  w_attn^T f16
  _Float16* wpt = (_Float16*)(ws + 14680064);   //  2 MB  w_proj^T f16
  _Float16* qh  = (_Float16*)(ws + 16777216);   //  8 MB  Q [B,H,S,hd]
  _Float16* kh  = (_Float16*)(ws + 25165824);   //  8 MB  K [B,H,S,hd]
  _Float16* vt  = (_Float16*)(ws + 33554432);   //  8 MB  V^T [B,H,hd,S]
  _Float16* ah  = (_Float16*)(ws + 41943040);   //  8 MB  attn out [B,S,D]

  k_cvt_f16<<<4096, 256, 0, stream>>>(x, xh, (int)(MM * DD));
  k_tcvt_f16<<<(3 * DD * DD + 255) / 256, 256, 0, stream>>>(w_attn, wat, DD, 3 * DD);
  k_tcvt_f16<<<(DD * DD + 255) / 256, 256, 0, stream>>>(w_proj, wpt, DD, DD);

  dim3 g1(3 * DD / 64, MM / 64);   // 48 x 64
  k_gemm_qkv<<<g1, 128, 0, stream>>>(xh, wat, b_attn, qh, kh, vt, out + A_ELEMS);

  k_attn<<<(BB * HH * (SS / 16)) / 4, 128, 0, stream>>>(qh, kh, vt, ah);

  dim3 g2(DD / 64, MM / 64);       // 16 x 64
  k_gemm_proj<<<g2, 128, 0, stream>>>(ah, wpt, b_proj, out);
}
